// RBFRNN_31722628448659
// MI455X (gfx1250) — matrix operand order repaired and checked by reading, not compiled, not run on hardware
//
#include <hip/hip_runtime.h>

// RBF-RNN ("grid cell" network) for MI455X / gfx1250.
//
// Pipeline (all on `stream`):
//   1. prep_frags      : W_rec (x2 copies), W_out f32 -> f16, pre-swizzled into WMMA
//                        B-fragment order. Two identical W_rec copies let the RNN
//                        alternate the streaming base with (t&1) so LICM cannot hoist
//                        the streamed fragment loads out of the time loop (which would
//                        blow the 512-VGPR budget and spill).
//   2. drive_kernel    : RBF speed + VonMises HD features, input GEMM -> driveT[t][u][b]
//                        in f32 so the RNN loads C-fragments straight into the WMMA
//                        accumulator registers (no f16->f32 cvt chain on the scan's
//                        critical path).
//   3. rnn_kernel      : ONE persistent workgroup, 8 waves (2/SIMD -> 512-VGPR budget).
//                        Per wave: 2 n16 tiles of W_rec pinned in VGPRs (256 regs),
//                        2 n16 tiles streamed from L2 once per step (256 KB/step);
//                        h ping-pongs in LDS; 256 WMMA/wave/step; 1 barrier/step.
//                        A-fragment pipelining is left to the scheduler (it provably
//                        hoists ds_loads under its pressure model; a manual double
//                        buffer pushed allocation over 512 regs and caused a spill).
//   4. out_gemm        : states @ W_out with WMMA, relu -> out[t][b][n] (f32)
//   5. centers_kernel  : sum over t + einsum('btn,btd->bnd') folded
//   6. final_kernel    : sum over n + einsum('btn,bnd->btd') folded

typedef __attribute__((ext_vector_type(16))) _Float16 v16h;
typedef __attribute__((ext_vector_type(8)))  _Float16 v8h;
typedef __attribute__((ext_vector_type(8)))  float    v8f;

constexpr int cB = 64;    // batch
constexpr int cT = 512;   // time steps
constexpr int cU = 512;   // units
constexpr int cN = 256;   // output nodes
constexpr int NSPD = 20;
constexpr int NHD  = 50;
constexpr int ROWP = 520; // LDS row stride in halves (multiple of 8; 520*2B = 1040B = 260 dwords,
                          // 260 % 64 = 4 -> consecutive rows shift 4 banks: conflict-free b128 reads)

// ---------------------------------------------------------------------------
// Fragment layout (16x16x32 f16 WMMA, wave32):
//   B (K=32 x N=16): lane L holds column N=L%16, j 0..7 -> K=(L/16)*8+j, j 8..15 -> K=16+(L/16)*8+j-8
//   A (M=16 x K=32): lane L holds row M=L%16, halves 0..7 -> K=(L/16)*8+0..7,
//                    halves 8..15 -> K=16+(L/16)*8+0..7
//   C/D f32        : lane L holds column N=L%16, vgpr v -> row M = 8*(L/16)+v
// frag buffer flat index: ((n16*16 + kb)*32 + lane)*16 + j
// ---------------------------------------------------------------------------
__global__ void prep_frags(const float* __restrict__ W, _Float16* __restrict__ frags,
                           int ncols, int total) {
  int idx = blockIdx.x * blockDim.x + threadIdx.x;
  if (idx >= total) return;
  int j    =  idx        & 15;
  int lane = (idx >> 4)  & 31;
  int kb   = (idx >> 9)  & 15;
  int n16  =  idx >> 13;
  int k = kb * 32 + ((lane >> 4) << 3) + (j & 7) + ((j >> 3) << 4);
  int n = n16 * 16 + (lane & 15);
  frags[idx] = (_Float16)W[(size_t)k * ncols + n];
}

// ---------------------------------------------------------------------------
// Features + input GEMM. One block per timestep t; writes driveT[t][u][b] f32.
// ---------------------------------------------------------------------------
__global__ __launch_bounds__(256)
void drive_kernel(const float* __restrict__ vel, const float* __restrict__ W_speed,
                  const float* __restrict__ W_hd, const float* __restrict__ b_rec,
                  float* __restrict__ driveT) {
  __shared__ float sfeat[cB][25];  // 20 used, padded (gcd(25,64)=1)
  __shared__ float hfeat[cB][53];  // 50 used, padded
  int t = blockIdx.x;
  int tid = threadIdx.x;
  if (tid < cB) {
    int b = tid;
    float s  = vel[((size_t)b * cT + t) * 2 + 0];
    float hd = vel[((size_t)b * cT + t) * 2 + 1];
    const float coef = -800.0f;                    // -0.5 / (0.025^2)
    #pragma unroll
    for (int i = 0; i < NSPD; i++) {
      float c = 0.5f * (float)i / 19.0f;           // linspace(0, 0.5, 20)
      float d = s - c;
      sfeat[b][i] = __expf(coef * d * d);
    }
    const float kappa  = 6.2831853071795865f;
    const float hdnorm = 0.9783932f;               // 1 / (2*pi * i0e(2*pi))
    #pragma unroll
    for (int i = 0; i < NHD; i++) {
      float c = kappa * (float)i / 49.0f;          // linspace(0, 2*pi, 50)
      hfeat[b][i] = __expf(kappa * (__cosf(hd - c) - 1.0f)) * hdnorm;
    }
  }
  __syncthreads();
  int b  = tid & 63;
  int c0 = tid >> 6;                               // 0..3
  for (int i = 0; i < 128; i++) {
    int u = i * 4 + c0;
    float acc = b_rec[u];
    #pragma unroll 4
    for (int k = 0; k < NSPD; k++) acc += sfeat[b][k] * W_speed[(size_t)k * cU + u];
    #pragma unroll 5
    for (int k = 0; k < NHD; k++)  acc += hfeat[b][k] * W_hd[(size_t)k * cU + u];
    driveT[((size_t)t * cU + u) * cB + b] = acc;
  }
}

// ---------------------------------------------------------------------------
// Persistent single-WGP RNN scan. 256 threads = 8 waves; wave w owns unit
// columns [64w, 64w+64) = n16 tiles {4w..4w+3}. Tiles 4w,4w+1 pinned in VGPRs;
// tiles 4w+2,4w+3 streamed from L2 once per step via a (t&1)-alternating base
// (two identical fragment copies) so the loads are not loop-invariant.
// kb-outer loop keeps all 16 accumulators live (128 VGPRs); C-fragments load
// straight from f32 driveT into the accumulator registers.
// ---------------------------------------------------------------------------
__global__ __launch_bounds__(256) __attribute__((amdgpu_waves_per_eu(2, 2)))
void rnn_kernel(const _Float16* __restrict__ wrec2,   // two identical fragment copies
                const float* __restrict__ driveT,
                const float* __restrict__ start_state,
                _Float16* __restrict__ states) {
  __shared__ __align__(16) _Float16 hbuf[2][cB * ROWP];
  int tid  = threadIdx.x;
  int w    = tid >> 5;
  int lane = tid & 31;
  int hi   = lane >> 4;   // 0/1
  int lo   = lane & 15;

  // h0 = broadcast(start_state) over batch, f32 -> f16 into buffer 0
  for (int u = tid; u < cU; u += 256) {
    _Float16 h = (_Float16)start_state[u];
    for (int b = 0; b < cB; b++) hbuf[0][b * ROWP + u] = h;
  }

  // Pin n16 tiles 4w, 4w+1 of W_rec in VGPRs (256 regs) from copy 0
  v16h bR[2][16];
  #pragma unroll
  for (int nt = 0; nt < 2; nt++) {
    #pragma unroll
    for (int kb = 0; kb < 16; kb++) {
      bR[nt][kb] = *(const v16h*)(wrec2 +
                    (((size_t)((4 * w + nt) * 16 + kb) * 32 + lane) * 16));
    }
  }
  __syncthreads();

  int cur = 0;
  #pragma unroll 1   // MUST stay rolled: unrolling would re-invariant the stream base
  for (int t = 0; t < cT; t++) {
    int nxt = cur ^ 1;
    const _Float16* hb = &hbuf[cur][0];
    _Float16*       hn = &hbuf[nxt][0];
    // (t&1)-alternating base: defeats LICM hoisting of the streamed B fragments
    const _Float16* wstream = wrec2 + (size_t)(t & 1) * (cU * cU);
    if (t + 1 < cT) {  // pull next step's drive slice (16 KB/wave, f32) toward L0/L2
      const float* pf = driveT + ((size_t)(t + 1) * cU + w * 64) * cB;
      __builtin_prefetch(pf + lane * 32, 0, 0);
      __builtin_prefetch(pf + 1024 + lane * 32, 0, 0);
      __builtin_prefetch(pf + 2048 + lane * 32, 0, 0);
      __builtin_prefetch(pf + 3072 + lane * 32, 0, 0);
    }

    // init all 16 accumulators from drive_t (C fragments, direct f32 loads)
    v8f acc[4][4];
    #pragma unroll
    for (int mt = 0; mt < 4; mt++) {
      int bb = mt * 16 + hi * 8;
      #pragma unroll
      for (int nt = 0; nt < 4; nt++) {
        int u = (4 * w + nt) * 16 + lo;
        acc[mt][nt] = *(const v8f*)(driveT + ((size_t)t * cU + u) * cB + bb);
      }
    }

    #pragma unroll
    for (int kb = 0; kb < 16; kb++) {
      // streamed B fragments (n16 tiles 4w+2, 4w+3), loaded once per step from L2
      v16h bS0 = *(const v16h*)(wstream +
                   (((size_t)((4 * w + 2) * 16 + kb) * 32 + lane) * 16));
      v16h bS1 = *(const v16h*)(wstream +
                   (((size_t)((4 * w + 3) * 16 + kb) * 32 + lane) * 16));
      #pragma unroll
      for (int mt = 0; mt < 4; mt++) {
        // A fragment from LDS; scheduler pipelines these ds_loads ahead on its own
        union AF { v16h v; v8h h8[2]; } a;
        int m  = mt * 16 + lo;
        int k1 = kb * 32 + hi * 8;
        a.h8[0] = *(const v8h*)(hb + m * ROWP + k1);
        a.h8[1] = *(const v8h*)(hb + m * ROWP + k1 + 16);
        acc[mt][0] = __builtin_amdgcn_wmma_f32_16x16x32_f16(
            false, a.v, false, bR[0][kb], (short)0, acc[mt][0], false, false);
        acc[mt][1] = __builtin_amdgcn_wmma_f32_16x16x32_f16(
            false, a.v, false, bR[1][kb], (short)0, acc[mt][1], false, false);
        acc[mt][2] = __builtin_amdgcn_wmma_f32_16x16x32_f16(
            false, a.v, false, bS0, (short)0, acc[mt][2], false, false);
        acc[mt][3] = __builtin_amdgcn_wmma_f32_16x16x32_f16(
            false, a.v, false, bS1, (short)0, acc[mt][3], false, false);
      }
    }

    // relu + f16 into next buffer
    #pragma unroll
    for (int mt = 0; mt < 4; mt++) {
      #pragma unroll
      for (int nt = 0; nt < 4; nt++) {
        int u = (4 * w + nt) * 16 + lo;
        #pragma unroll
        for (int v = 0; v < 8; v++) {
          float f = acc[mt][nt][v];
          f = f > 0.0f ? f : 0.0f;
          int m = mt * 16 + hi * 8 + v;
          hn[m * ROWP + u] = (_Float16)f;
        }
      }
    }
    __syncthreads();
    // cooperative coalesced dump of h_{t+1} -> states[t][b][u] (f16)
    {
      int b     = tid >> 2;          // 0..63
      int chunk = (tid & 3) * 128;   // 0,128,256,384
      #pragma unroll
      for (int i = 0; i < 16; i++) {
        v8h val = *(const v8h*)(&hbuf[nxt][b * ROWP + chunk + i * 8]);
        *(v8h*)(states + ((size_t)t * cB + b) * cU + chunk + i * 8) = val;
      }
    }
    cur = nxt;
  }
}

// ---------------------------------------------------------------------------
// out = relu(states @ W_out + b_out). M=32768 (=t*64+b), N=256, K=512, WMMA.
// Block = 8 waves: 2 M-tiles x 4 N-groups (each N-group = 4 n16 tiles = 64 cols).
// ---------------------------------------------------------------------------
__global__ __launch_bounds__(256)
void out_gemm(const _Float16* __restrict__ states, const _Float16* __restrict__ wout_frags,
              const float* __restrict__ b_out, float* __restrict__ out) {
  int tid  = threadIdx.x;
  int w    = tid >> 5;
  int lane = tid & 31;
  int hi = lane >> 4, lo = lane & 15;
  int mt = w >> 2;
  int ng = w & 3;
  int mbase = blockIdx.x * 32 + mt * 16;

  v8f acc[4];
  #pragma unroll
  for (int i = 0; i < 4; i++) {
    float bo = b_out[(ng * 4 + i) * 16 + lo];
    #pragma unroll
    for (int v = 0; v < 8; v++) acc[i][v] = bo;
  }
  for (int kb = 0; kb < 16; kb++) {
    union { v16h v; v8h h8[2]; } af;
    int m  = mbase + lo;
    int k1 = kb * 32 + hi * 8;
    af.h8[0] = *(const v8h*)(states + (size_t)m * cU + k1);
    af.h8[1] = *(const v8h*)(states + (size_t)m * cU + k1 + 16);
    #pragma unroll
    for (int i = 0; i < 4; i++) {
      int n16 = ng * 4 + i;
      v16h bf = *(const v16h*)(wout_frags + (((size_t)(n16 * 16 + kb) * 32 + lane) * 16));
      acc[i] = __builtin_amdgcn_wmma_f32_16x16x32_f16(
          false, af.v, false, bf, (short)0, acc[i], false, false);
    }
  }
  #pragma unroll
  for (int i = 0; i < 4; i++) {
    int n = (ng * 4 + i) * 16 + lo;
    #pragma unroll
    for (int v = 0; v < 8; v++) {
      float f = acc[i][v];
      f = f > 0.0f ? f : 0.0f;
      out[((size_t)(mbase + hi * 8 + v)) * cN + n] = f;
    }
  }
}

// centers[b][n][d] = (sum_t out*r) / (sum_t out + T*1e-10)
__global__ void centers_kernel(const float* __restrict__ out, const float* __restrict__ r,
                               float* __restrict__ centers) {
  int gid = blockIdx.x * blockDim.x + threadIdx.x;  // b*256 + n
  int b = gid >> 8, n = gid & 255;
  float s = 0.f, s0 = 0.f, s1 = 0.f;
  for (int t = 0; t < cT; t++) {
    float o  = out[((size_t)t * cB + b) * cN + n];
    float r0 = r[((size_t)b * cT + t) * 2 + 0];
    float r1 = r[((size_t)b * cT + t) * 2 + 1];
    s += o; s0 += o * r0; s1 += o * r1;
  }
  float denom = s + (float)cT * 1e-10f;
  centers[((size_t)b * cN + n) * 2 + 0] = s0 / denom;
  centers[((size_t)b * cN + n) * 2 + 1] = s1 / denom;
}

// pos[b][t][d] = (sum_n out*centers) / (sum_n out + N*1e-10)
__global__ __launch_bounds__(256)
void final_kernel(const float* __restrict__ out, const float* __restrict__ centers,
                  float* __restrict__ pos) {
  __shared__ float c0[cN], c1[cN];
  int gid = blockIdx.x * blockDim.x + threadIdx.x;  // b*512 + t (b constant per block)
  int b = gid >> 9, t = gid & 511;
  int tid = threadIdx.x;
  c0[tid] = centers[((size_t)b * cN + tid) * 2 + 0];
  c1[tid] = centers[((size_t)b * cN + tid) * 2 + 1];
  __syncthreads();
  float sn = 0.f, p0 = 0.f, p1 = 0.f;
  for (int n = 0; n < cN; n++) {
    float o = out[((size_t)t * cB + b) * cN + n];
    sn += o; p0 += o * c0[n]; p1 += o * c1[n];
  }
  float denom = sn + (float)cN * 1e-10f;
  pos[((size_t)b * cT + t) * 2 + 0] = p0 / denom;
  pos[((size_t)b * cT + t) * 2 + 1] = p1 / denom;
}

extern "C" void kernel_launch(void* const* d_in, const int* in_sizes, int n_in,
                              void* d_out, int out_size, void* d_ws, size_t ws_size,
                              hipStream_t stream) {
  const float* vel         = (const float*)d_in[0];
  const float* r           = (const float*)d_in[1];
  const float* start_state = (const float*)d_in[2];
  const float* W_speed     = (const float*)d_in[3];
  const float* W_hd        = (const float*)d_in[4];
  const float* W_rec       = (const float*)d_in[5];
  const float* b_rec       = (const float*)d_in[6];
  const float* W_out       = (const float*)d_in[7];
  const float* b_out       = (const float*)d_in[8];
  float* pos = (float*)d_out;

  char* ws = (char*)d_ws;
  size_t off = 0;
  auto alloc = [&](size_t bytes) -> void* {
    void* p = ws + off;
    off = (off + bytes + 255) & ~(size_t)255;
    return p;
  };
  _Float16* wrec2      = (_Float16*)alloc((size_t)2 * cU * cU * 2);    // 1 MB (two copies)
  _Float16* wout_frags = (_Float16*)alloc((size_t)cU * cN * 2);        // 256 KB
  float*    driveT     = (float*)alloc((size_t)cT * cU * cB * 4);      // 64 MB
  _Float16* states     = (_Float16*)alloc((size_t)cT * cB * cU * 2);   // 32 MB
  float*    out        = (float*)alloc((size_t)cT * cB * cN * 4);      // 32 MB
  float*    centers    = (float*)alloc((size_t)cB * cN * 2 * 4);       // 128 KB
  (void)ws_size; (void)in_sizes; (void)n_in; (void)out_size;

  prep_frags<<<(cU * cU) / 256, 256, 0, stream>>>(W_rec, wrec2, cU, cU * cU);
  prep_frags<<<(cU * cU) / 256, 256, 0, stream>>>(W_rec, wrec2 + (size_t)cU * cU, cU, cU * cU);
  prep_frags<<<(cU * cN) / 256, 256, 0, stream>>>(W_out, wout_frags, cN, cU * cN);
  drive_kernel<<<cT, 256, 0, stream>>>(vel, W_speed, W_hd, b_rec, driveT);
  rnn_kernel<<<1, 256, 0, stream>>>(wrec2, driveT, start_state, states);
  out_gemm<<<(cT * cB) / 32, 256, 0, stream>>>(states, wout_frags, b_out, out);
  centers_kernel<<<(cB * cN) / 256, 256, 0, stream>>>(out, r, centers);
  final_kernel<<<(cB * cT) / 256, 256, 0, stream>>>(out, centers, pos);
}
